// ConvFlow_3951369912645
// MI455X (gfx1250) — matrix operand order repaired and checked by reading, not compiled, not run on hardware
//
#include <hip/hip_runtime.h>
#include <cmath>

typedef __attribute__((ext_vector_type(16))) __bf16 v16bf;
typedef __attribute__((ext_vector_type(8)))  float  v8f;

// Problem constants (from the reference)
constexpr int BN    = 16;
constexpr int CH    = 192;
constexpr int TT    = 1024;
constexpr int HALFC = 96;
constexpr int FILT  = 256;
constexpr int NBINS = 10;
constexpr int NCOL  = BN * TT;                 // 16384
constexpr int MPROJ = HALFC * (3 * NBINS - 1); // 2784
constexpr int MPROJ_PAD = 2816;                // padded to multiple of 64
constexpr float TBND = 5.0f;
constexpr float MINW = 0.001f;
constexpr float MINH = 0.001f;
constexpr float MIND = 0.001f;
constexpr float INV_SQRT_FILT = 0.0625f;       // 1/sqrt(256)

constexpr int KPAD = 40; // LDS k-stride (bf16 elems): conflict-free b128 frag loads

// Workspace layout (bytes), all offsets 256B-aligned
constexpr size_t WPRE_OFF  = 0;                                    // 256*96 bf16
constexpr size_t WC11_OFF  = 49152;                                // 3*256*256 bf16
constexpr size_t WPROJ_OFF = WC11_OFF + 3*FILT*FILT*2;             // 2816*256 bf16 (padded)
constexpr size_t PB_OFF    = WPROJ_OFF + (size_t)MPROJ_PAD*FILT*2; // 2816 fp32 padded bias
constexpr size_t XBF_OFF   = PB_OFF + (size_t)MPROJ_PAD*4;         // bf16 [16384][256] K-major
constexpr size_t H_OFF     = XBF_OFF + (size_t)FILT*NCOL*2;        // fp32 [256][16384]
constexpr size_t Y_OFF     = H_OFF + (size_t)FILT*NCOL*4;          // fp32 [256][16384]
constexpr size_t P_OFF     = Y_OFF + (size_t)FILT*NCOL*4;          // fp32 [2816][1024] per batch
constexpr size_t PART_OFF  = P_OFF + (size_t)MPROJ_PAD*TT*4;       // fp32 [16][384] partials

// ---------------- elementwise helpers ----------------

// dst[i] = bf16(src[i]) for i < nsrc, else 0   (zero-pads weight tails)
__global__ void cvt_pad_bf16(const float* __restrict__ s, __bf16* __restrict__ d,
                             int nsrc, int ntot) {
  int i = blockIdx.x * 256 + threadIdx.x;
  if (i < ntot) d[i] = (i < nsrc) ? (__bf16)s[i] : (__bf16)0.0f;
}

__global__ void pad_bias_f32(const float* __restrict__ s, float* __restrict__ d,
                             int nsrc, int ntot) {
  int i = blockIdx.x * 256 + threadIdx.x;
  if (i < ntot) d[i] = (i < nsrc) ? s[i] : 0.0f;
}

// Activation panel is K-major: X[n*Kld + c].
// pack_x0: X[n*96 + c] = bf16(x[b, c, t]), n = b*1024 + t
__global__ void pack_x0(const float* __restrict__ x, __bf16* __restrict__ X) {
  int i = blockIdx.x * 256 + threadIdx.x;
  if (i >= HALFC * NCOL) return;
  int c = i % HALFC, n = i / HALFC, b = n >> 10, t = n & 1023;
  X[i] = (__bf16)x[(size_t)b * CH * TT + (size_t)c * TT + t];
}

// X[n*256 + c] = bf16(H[c][n] * mask[n])
__global__ void pack_h(const float* __restrict__ H, const float* __restrict__ mask,
                       __bf16* __restrict__ X) {
  int i = blockIdx.x * 256 + threadIdx.x;
  if (i >= FILT * NCOL) return;
  int c = i % FILT, n = i / FILT;
  X[i] = (__bf16)(H[(size_t)c * NCOL + n] * mask[n]);
}

// out[b, c, t] = x[b, c, t] * mask, c < 96 (pass-through half)
__global__ void copy_x0(const float* __restrict__ x, const float* __restrict__ mask,
                        float* __restrict__ out) {
  int i = blockIdx.x * 256 + threadIdx.x;
  if (i >= BN * HALFC * TT) return;
  int b = i / (HALFC * TT), r = i % (HALFC * TT), c = r / TT, t = r % TT;
  size_t o = (size_t)b * CH * TT + (size_t)c * TT + t;
  out[o] = x[o] * mask[b * TT + t];
}

// Depthwise conv, kernel 3, dilation dil, zero-pad within each batch's T segment
__global__ void dwconv(const float* __restrict__ H, const float* __restrict__ mask,
                       const float* __restrict__ w, const float* __restrict__ bias,
                       float* __restrict__ Y, int dil) {
  int i = blockIdx.x * 256 + threadIdx.x;
  if (i >= FILT * NCOL) return;
  int o = i / NCOL, n = i % NCOL, b = n >> 10, t = n & 1023;
  float acc = bias[o];
#pragma unroll
  for (int j = 0; j < 3; ++j) {
    int tt = t + (j - 1) * dil;
    if (tt >= 0 && tt < TT) {
      int nn = (b << 10) + tt;
      acc += w[o * 3 + j] * H[(size_t)o * NCOL + nn] * mask[nn];
    }
  }
  Y[i] = acc;
}

// LayerNorm over 256 channels per column + exact GELU.
// mode 0: write K-major bf16 panel Xout[n*256 + c]; mode 1: H += gelu(LN(Y))
__global__ __launch_bounds__(256) void ln_gelu(const float* __restrict__ Y,
                                               const float* __restrict__ g,
                                               const float* __restrict__ bb,
                                               float* __restrict__ Hres,
                                               __bf16* __restrict__ Xout, int mode) {
  const int lane = threadIdx.x;               // 0..31 (wave32)
  const int n = blockIdx.x * 8 + threadIdx.y; // column
  float vals[8];
  float s = 0.f, ss = 0.f;
#pragma unroll
  for (int j = 0; j < 8; ++j) {
    int c = lane + 32 * j;
    float v = Y[(size_t)c * NCOL + n];
    vals[j] = v; s += v; ss += v * v;
  }
#pragma unroll
  for (int off = 16; off > 0; off >>= 1) {
    s  += __shfl_xor(s,  off, 32);
    ss += __shfl_xor(ss, off, 32);
  }
  float mean = s * (1.0f / 256.0f);
  float var  = ss * (1.0f / 256.0f) - mean * mean;
  float rs   = rsqrtf(var + 1e-5f);
#pragma unroll
  for (int j = 0; j < 8; ++j) {
    int c = lane + 32 * j;
    float v  = (vals[j] - mean) * rs * g[c] + bb[c];
    float gl = 0.5f * v * (1.0f + erff(v * 0.70710678118654752f));
    if (mode == 0) Xout[(size_t)n * FILT + c] = (__bf16)gl;   // lane-coalesced
    else           Hres[(size_t)c * NCOL + n] += gl;
  }
}

// ---------------- WMMA bf16 GEMM ----------------
// Out[m, n] = bias[m] + sum_k W[m*K + k] * X[n*K + k]   (X panel is K-major)
// Block tile 64(M) x 256(N); 8 waves in a 2x4 grid; each wave owns a 32x64 tile:
// 2 A-frags x 4 B-frags -> 8 WMMAs per k-step against 12 ds_load_b128 (1:1.5),
// 4x fewer barriers per WMMA than a 16x32 wave tile. M is always a multiple
// of 64 (proj weights/bias zero-padded to 2816).
__global__ __launch_bounds__(256) void gemm_bf16(const __bf16* __restrict__ W,
                                                 const __bf16* __restrict__ X,
                                                 const float* __restrict__ bias,
                                                 float* __restrict__ Out,
                                                 int M, int N, int K, int ldo) {
  __shared__ alignas(16) __bf16 As[64 * KPAD];
  __shared__ alignas(16) __bf16 Bs[256 * KPAD];
  const int tid  = threadIdx.x;
  const int lane = tid & 31, wave = tid >> 5;
  const int wm = wave & 1, wn = wave >> 1;      // 2(M) x 4(N) wave grid
  const int m0 = blockIdx.x * 64, n0 = blockIdx.y * 256;

  union CF { v8f v; float e[8]; };
  CF acc[2][4];
#pragma unroll
  for (int i = 0; i < 2; ++i)
#pragma unroll
    for (int j = 0; j < 4; ++j)
#pragma unroll
      for (int r = 0; r < 8; ++r) acc[i][j].e[r] = 0.f;

  const int ar = tid >> 2, ac = (tid & 3) * 8;  // A stage: 8 bf16 (16B) / thread
  const __bf16* wrow = &W[(size_t)(m0 + ar) * K + ac];
  const __bf16* xrow = &X[(size_t)(n0 + tid) * K]; // B stage: full k-row for n=tid

  for (int k0 = 0; k0 < K; k0 += 32) {
    *(uint4*)&As[ar * KPAD + ac] = *(const uint4*)(wrow + k0);
#pragma unroll
    for (int q = 0; q < 4; ++q)
      *(uint4*)&Bs[tid * KPAD + q * 8] = *(const uint4*)(xrow + k0 + q * 8);
    if (k0 + 32 < K) { // global_prefetch_b8 for next K tile
      __builtin_prefetch(wrow + k0 + 32, 0, 1);
      __builtin_prefetch(xrow + k0 + 32, 0, 1);
    }
    __syncthreads();

    union ABu { v16bf v; uint4 q[2]; };
    ABu a[2], bfr[4];
    const int half = lane >> 4, l16 = lane & 15;
#pragma unroll
    for (int i = 0; i < 2; ++i) { // A frag: e[0..7]=k half*8.., e[8..15]=k 16+half*8..
      const __bf16* arow = &As[(wm * 32 + i * 16 + l16) * KPAD + half * 8];
      a[i].q[0] = *(const uint4*)(arow);
      a[i].q[1] = *(const uint4*)(arow + 16);
    }
#pragma unroll
    for (int j = 0; j < 4; ++j) { // B frag: e[0..7]=k half*8.., e[8..15]=k 16+half*8.. for column n
      const __bf16* brow = &Bs[(wn * 64 + j * 16 + l16) * KPAD + half * 8];
      bfr[j].q[0] = *(const uint4*)(brow);
      bfr[j].q[1] = *(const uint4*)(brow + 16);
    }
#pragma unroll
    for (int i = 0; i < 2; ++i)
#pragma unroll
      for (int j = 0; j < 4; ++j)
        acc[i][j].v = __builtin_amdgcn_wmma_f32_16x16x32_bf16(
            false, a[i].v, false, bfr[j].v, (short)0, acc[i][j].v, false, false);
    __syncthreads();
  }

  const int half = lane >> 4, l16 = lane & 15;
#pragma unroll
  for (int i = 0; i < 2; ++i) {
#pragma unroll
    for (int r = 0; r < 8; ++r) { // C/D layout: VGPR r -> M = half*8 + r, N = l16
      int m = m0 + wm * 32 + i * 16 + half * 8 + r;
      float bv = bias[m];
#pragma unroll
      for (int j = 0; j < 4; ++j)
        Out[(size_t)m * ldo + n0 + wn * 64 + j * 16 + l16] = acc[i][j].e[r] + bv;
    }
  }
}

// ---------------- Rational-quadratic spline ----------------
// One batch at a time; P is [2816][1024] fp32 for batch b (rows >= 2784 unused).
__global__ __launch_bounds__(256) void spline(const float* __restrict__ P,
                                              const float* __restrict__ x,
                                              const float* __restrict__ mask,
                                              float* __restrict__ out,
                                              float* __restrict__ partial, int b) {
  int idx = blockIdx.x * 256 + threadIdx.x; // < 96*1024
  int c = idx >> 10, t = idx & 1023;
  float mv = mask[(b << 10) + t];
  float p[29];
#pragma unroll
  for (int j = 0; j < 29; ++j) p[j] = P[(size_t)(c * 29 + j) * TT + t] * mv;

  float w[10], h[10];
  { // softmax widths
    float mx = -1e30f;
#pragma unroll
    for (int j = 0; j < 10; ++j) { w[j] = p[j] * INV_SQRT_FILT; mx = fmaxf(mx, w[j]); }
    float sum = 0.f;
#pragma unroll
    for (int j = 0; j < 10; ++j) { w[j] = expf(w[j] - mx); sum += w[j]; }
    float inv = 1.0f / sum;
#pragma unroll
    for (int j = 0; j < 10; ++j) w[j] = MINW + (1.0f - MINW * 10.0f) * w[j] * inv;
  }
  { // softmax heights
    float mx = -1e30f;
#pragma unroll
    for (int j = 0; j < 10; ++j) { h[j] = p[10 + j] * INV_SQRT_FILT; mx = fmaxf(mx, h[j]); }
    float sum = 0.f;
#pragma unroll
    for (int j = 0; j < 10; ++j) { h[j] = expf(h[j] - mx); sum += h[j]; }
    float inv = 1.0f / sum;
#pragma unroll
    for (int j = 0; j < 10; ++j) h[j] = MINH + (1.0f - MINH * 10.0f) * h[j] * inv;
  }
  float cw[11], chh[11];
  cw[0] = -TBND; chh[0] = -TBND;
  float aw = 0.f, ah = 0.f;
#pragma unroll
  for (int j = 0; j < 9; ++j) {
    aw += w[j]; cw[j + 1]  = -TBND + 2.0f * TBND * aw;
    ah += h[j]; chh[j + 1] = -TBND + 2.0f * TBND * ah;
  }
  cw[10] = TBND; chh[10] = TBND;

  float d[11];
  d[0] = 1.0f; d[10] = 1.0f; // MIND + softplus(DERIV_CONST) == 1 exactly
#pragma unroll
  for (int j = 1; j < 10; ++j) {
    float u = p[20 + j - 1];
    float sp = (u > 20.f) ? u : log1pf(expf(u));
    d[j] = MIND + sp;
  }

  float xv = x[(size_t)b * CH * TT + (size_t)(HALFC + c) * TT + t];
  bool inside = (xv >= -TBND) && (xv <= TBND);
  float xc = fminf(fmaxf(xv, -TBND), TBND);

  int bin = 0;
#pragma unroll
  for (int j = 1; j < 10; ++j) if (xc >= cw[j]) bin = j;

  float icw = 0.f, iw = 1.f, ich = 0.f, ih = 1.f, ider = 1.f, iderp1 = 1.f;
#pragma unroll
  for (int j = 0; j < 10; ++j) {
    if (j == bin) {
      icw = cw[j]; iw = cw[j + 1] - cw[j];
      ich = chh[j]; ih = chh[j + 1] - chh[j];
      ider = d[j]; iderp1 = d[j + 1];
    }
  }
  float idl = ih / iw;
  float th = (xc - icw) / iw;
  float ttv = th * (1.0f - th);
  float num = ih * (idl * th * th + ider * ttv);
  float den = idl + (ider + iderp1 - 2.0f * idl) * ttv;
  float outv = ich + num / den;
  float omt = 1.0f - th;
  float dnum = idl * idl * (iderp1 * th * th + 2.0f * idl * ttv + ider * omt * omt);
  float lad = logf(dnum) - 2.0f * logf(den);
  if (!inside) { outv = xv; lad = 0.0f; }

  out[(size_t)b * CH * TT + (size_t)(HALFC + c) * TT + t] = outv * mv;

  __shared__ float red[256];
  red[threadIdx.x] = lad * mv;
  __syncthreads();
#pragma unroll
  for (int s2 = 128; s2 > 0; s2 >>= 1) {
    if (threadIdx.x < s2) red[threadIdx.x] += red[threadIdx.x + s2];
    __syncthreads();
  }
  if (threadIdx.x == 0) partial[blockIdx.x] = red[0];
}

// Deterministic logdet reduction: fixed summation order over 384 partials
__global__ __launch_bounds__(256) void reduce_logdet(const float* __restrict__ partial,
                                                     float* __restrict__ ld) {
  int b = blockIdx.x;
  __shared__ float red[256];
  float s = 0.f;
  for (int i = threadIdx.x; i < 384; i += 256) s += partial[b * 384 + i];
  red[threadIdx.x] = s;
  __syncthreads();
#pragma unroll
  for (int s2 = 128; s2 > 0; s2 >>= 1) {
    if (threadIdx.x < s2) red[threadIdx.x] += red[threadIdx.x + s2];
    __syncthreads();
  }
  if (threadIdx.x == 0) ld[b] = red[0];
}

// ---------------- host launcher ----------------

extern "C" void kernel_launch(void* const* d_in, const int* in_sizes, int n_in,
                              void* d_out, int out_size, void* d_ws, size_t ws_size,
                              hipStream_t stream) {
  (void)in_sizes; (void)n_in; (void)out_size; (void)ws_size;
  const float* x      = (const float*)d_in[0];
  const float* mask   = (const float*)d_in[1];
  const float* pre_w  = (const float*)d_in[2];
  const float* pre_b  = (const float*)d_in[3];
  const float* sep_w  = (const float*)d_in[4];
  const float* sep_b  = (const float*)d_in[5];
  const float* n1_g   = (const float*)d_in[6];
  const float* n1_b   = (const float*)d_in[7];
  const float* c11_w  = (const float*)d_in[8];
  const float* c11_b  = (const float*)d_in[9];
  const float* n2_g   = (const float*)d_in[10];
  const float* n2_b   = (const float*)d_in[11];
  const float* proj_w = (const float*)d_in[12];
  const float* proj_b = (const float*)d_in[13];

  char* ws = (char*)d_ws;
  __bf16* WpreB  = (__bf16*)(ws + WPRE_OFF);
  __bf16* Wc11B  = (__bf16*)(ws + WC11_OFF);
  __bf16* WprojB = (__bf16*)(ws + WPROJ_OFF);
  float*  PBpad  = (float*)(ws + PB_OFF);
  __bf16* Xbf    = (__bf16*)(ws + XBF_OFF);
  float*  H      = (float*)(ws + H_OFF);
  float*  Y      = (float*)(ws + Y_OFF);
  float*  P      = (float*)(ws + P_OFF);
  float*  part   = (float*)(ws + PART_OFF);
  float*  outp   = (float*)d_out;
  float*  ld     = outp + (size_t)BN * CH * TT;

  cvt_pad_bf16<<<(FILT * HALFC + 255) / 256, 256, 0, stream>>>(
      pre_w, WpreB, FILT * HALFC, FILT * HALFC);
  cvt_pad_bf16<<<(3 * FILT * FILT + 255) / 256, 256, 0, stream>>>(
      c11_w, Wc11B, 3 * FILT * FILT, 3 * FILT * FILT);
  cvt_pad_bf16<<<(MPROJ_PAD * FILT + 255) / 256, 256, 0, stream>>>(
      proj_w, WprojB, MPROJ * FILT, MPROJ_PAD * FILT);
  pad_bias_f32<<<(MPROJ_PAD + 255) / 256, 256, 0, stream>>>(
      proj_b, PBpad, MPROJ, MPROJ_PAD);

  pack_x0<<<(HALFC * NCOL) / 256, 256, 0, stream>>>(x, Xbf);
  // h = pre_w @ x0 + pre_b    (M=256, K=96, N=16384)
  gemm_bf16<<<dim3(FILT / 64, NCOL / 256), 256, 0, stream>>>(
      WpreB, Xbf, pre_b, H, FILT, NCOL, HALFC, NCOL);

  int dil = 1;
  for (int i = 0; i < 3; ++i) {
    dwconv<<<(FILT * NCOL) / 256, 256, 0, stream>>>(
        H, mask, sep_w + i * FILT * 3, sep_b + i * FILT, Y, dil);
    ln_gelu<<<NCOL / 8, dim3(32, 8), 0, stream>>>(
        Y, n1_g + i * FILT, n1_b + i * FILT, nullptr, Xbf, 0);
    // y = c11_w[i] @ gelu(ln(y)) + c11_b[i]   (M=256, K=256)
    gemm_bf16<<<dim3(FILT / 64, NCOL / 256), 256, 0, stream>>>(
        Wc11B + (size_t)i * FILT * FILT, Xbf, c11_b + i * FILT, Y,
        FILT, NCOL, FILT, NCOL);
    ln_gelu<<<NCOL / 8, dim3(32, 8), 0, stream>>>(
        Y, n2_g + i * FILT, n2_b + i * FILT, H, nullptr, 1);
    dil *= 3;
  }

  pack_h<<<(FILT * NCOL) / 256, 256, 0, stream>>>(H, mask, Xbf);
  copy_x0<<<(BN * HALFC * TT) / 256, 256, 0, stream>>>(x, mask, outp);

  // proj GEMM per batch (M=2816 padded, N=1024), fused with spline consume
  for (int b = 0; b < BN; ++b) {
    gemm_bf16<<<dim3(MPROJ_PAD / 64, TT / 256), 256, 0, stream>>>(
        WprojB, Xbf + (size_t)b * TT * FILT, PBpad, P, MPROJ_PAD, TT, FILT, TT);
    spline<<<(HALFC * TT) / 256, 256, 0, stream>>>(P, x, mask, outp, part + b * 384, b);
  }
  reduce_logdet<<<BN, 256, 0, stream>>>(part, ld);
}
